// Qwen3_5VisionAttention_26431228739842
// MI455X (gfx1250) — matrix operand order repaired and checked by reading, not compiled, not run on hardware
//
#include <hip/hip_runtime.h>
#include <hip/hip_bf16.h>
#include <stdint.h>

#define SEQ 4096
#define HIDDEN 1280
#define NHEADS 16
#define HDIM 80
#define HDIM_PAD 96
#define QKV_N 3840
static __device__ __constant__ float INV_SCALE = 0.11180339887498948f; // 1/sqrt(80)

typedef _Float16 h16 __attribute__((ext_vector_type(16)));
typedef _Float16 h8  __attribute__((ext_vector_type(8)));
typedef float    f8  __attribute__((ext_vector_type(8)));

// ---------------------------------------------------------------------------
// Fragment helpers for v_wmma_f32_16x16x32_f16 (wave32).
// A (16x32, row-major source, leading dim ld):
//   lane l (half = l>>4, r = l&15) holds row r;
//   elements 0..7  = K (l<16 ? 0..7  : 16..23)
//   elements 8..15 = K (l<16 ? 8..15 : 24..31)
// B (32x16): source stored K-contiguous per column (B^T row-major, ld = K-stride):
//   lane l holds column n = l&15; element i -> K = (l>>4)*16 + i  (one contiguous
//   32-byte run of 16 halves).
// C/D (16x16 f32): lane l -> column n = l&15; element r -> row r + 8*(l>>4).
// ---------------------------------------------------------------------------
__device__ __forceinline__ h16 load_a_frag(const _Float16* p, int ld) {
  int lane = threadIdx.x & 31;
  int hf = lane >> 4, r = lane & 15;
  const _Float16* q = p + (size_t)r * ld + hf * 16;
  h8 lo = *(const h8*)(q);
  h8 hi = *(const h8*)(q + 8);
  return __builtin_shufflevector(lo, hi, 0,1,2,3,4,5,6,7,8,9,10,11,12,13,14,15);
}

__device__ __forceinline__ h16 load_b_frag(const _Float16* p, int ld) {
  int lane = threadIdx.x & 31;
  int hf = lane >> 4, n = lane & 15;
  return *(const h16*)(p + (size_t)n * ld + hf * 16);
}

__device__ __forceinline__ f8 wmma_f16(h16 a, h16 b, f8 c) {
  return __builtin_amdgcn_wmma_f32_16x16x32_f16(false, a, false, b, (short)0, c,
                                                false, false);
}

// CDNA5 async DMA: global -> LDS, 16 bytes per lane, tracked on ASYNCcnt.
__device__ __forceinline__ void async_copy_b128(uint32_t lds_off, const void* g) {
  asm volatile("global_load_async_to_lds_b128 %0, %1, off"
               :: "v"(lds_off), "v"((uint64_t)(uintptr_t)g)
               : "memory");
}
__device__ __forceinline__ void wait_async0() {
  asm volatile("s_wait_asynccnt 0" ::: "memory");
}

// ---------------------------------------------------------------------------
// 1) fp32 -> f16 elementwise
// ---------------------------------------------------------------------------
__global__ void cvt_f32_to_f16_kernel(const float* __restrict__ src,
                                      _Float16* __restrict__ dst, int n) {
  int i = blockIdx.x * blockDim.x + threadIdx.x;
  for (; i < n; i += gridDim.x * blockDim.x) dst[i] = (_Float16)src[i];
}

// 2) fp32 (K x N row-major) -> f16 transposed (N x K row-major)
__global__ void transpose_to_f16_kernel(const float* __restrict__ w,
                                        _Float16* __restrict__ wT, int K, int N) {
  int i = blockIdx.x * blockDim.x + threadIdx.x;
  int total = K * N;
  for (; i < total; i += gridDim.x * blockDim.x) {
    int k = i / N, n = i - k * N;
    wT[(size_t)n * K + k] = (_Float16)w[i];
  }
}

// ---------------------------------------------------------------------------
// 3) GEMM core: C[M x N] = A[M x K] * B^T-layout[N x K], f16 in, register
//    double-buffered. 128 threads = 4 waves; each wave: 32 rows x 64 cols
//    (2 A-frags x 4 B-frags -> 8 wmma per K-step). Block tile: 128 x 64.
// ---------------------------------------------------------------------------
template <bool F32OUT>
__device__ __forceinline__ void gemm_tile(const _Float16* __restrict__ A,
                                          const _Float16* __restrict__ BT,
                                          int ldk, int ncols,
                                          _Float16* __restrict__ outh,
                                          float* __restrict__ outf,
                                          const float* __restrict__ bias) {
  int w = threadIdx.x >> 5;
  int lane = threadIdx.x & 31;
  int hf = lane >> 4, n = lane & 15;
  int m0 = blockIdx.y * 128 + w * 32;
  int n0 = blockIdx.x * 64;

  const _Float16* ap0 = A + (size_t)m0 * ldk;
  const _Float16* ap1 = ap0 + (size_t)16 * ldk;

  f8 acc[2][4] = {{f8{}, f8{}, f8{}, f8{}}, {f8{}, f8{}, f8{}, f8{}}};
  h16 a0 = load_a_frag(ap0, ldk);
  h16 a1 = load_a_frag(ap1, ldk);
  h16 b0 = load_b_frag(BT + (size_t)(n0 + 0)  * ldk, ldk);
  h16 b1 = load_b_frag(BT + (size_t)(n0 + 16) * ldk, ldk);
  h16 b2 = load_b_frag(BT + (size_t)(n0 + 32) * ldk, ldk);
  h16 b3 = load_b_frag(BT + (size_t)(n0 + 48) * ldk, ldk);

  for (int k0 = 0; k0 < ldk; k0 += 32) {
    int kn = (k0 + 32 < ldk) ? k0 + 32 : k0;   // clamped prefetch (no OOB)
    __builtin_prefetch(ap0 + kn + 32, 0, 0);
    h16 na0 = load_a_frag(ap0 + kn, ldk);
    h16 na1 = load_a_frag(ap1 + kn, ldk);
    h16 nb0 = load_b_frag(BT + (size_t)(n0 + 0)  * ldk + kn, ldk);
    h16 nb1 = load_b_frag(BT + (size_t)(n0 + 16) * ldk + kn, ldk);
    h16 nb2 = load_b_frag(BT + (size_t)(n0 + 32) * ldk + kn, ldk);
    h16 nb3 = load_b_frag(BT + (size_t)(n0 + 48) * ldk + kn, ldk);

    acc[0][0] = wmma_f16(a0, b0, acc[0][0]);
    acc[1][0] = wmma_f16(a1, b0, acc[1][0]);
    acc[0][1] = wmma_f16(a0, b1, acc[0][1]);
    acc[1][1] = wmma_f16(a1, b1, acc[1][1]);
    acc[0][2] = wmma_f16(a0, b2, acc[0][2]);
    acc[1][2] = wmma_f16(a1, b2, acc[1][2]);
    acc[0][3] = wmma_f16(a0, b3, acc[0][3]);
    acc[1][3] = wmma_f16(a1, b3, acc[1][3]);

    a0 = na0; a1 = na1; b0 = nb0; b1 = nb1; b2 = nb2; b3 = nb3;
  }

#pragma unroll
  for (int g = 0; g < 2; ++g)
#pragma unroll
    for (int u = 0; u < 4; ++u)
#pragma unroll
      for (int r = 0; r < 8; ++r) {
        int row = m0 + g * 16 + r + 8 * hf;
        int col = n0 + 16 * u + n;
        if (F32OUT)
          outf[(size_t)row * ncols + col] = acc[g][u][r] + bias[col];
        else
          outh[(size_t)row * ncols + col] = (_Float16)acc[g][u][r];
      }
}

__global__ void qkv_gemm_kernel(const _Float16* __restrict__ xh,
                                const _Float16* __restrict__ wT,
                                _Float16* __restrict__ qkv_h) {
  gemm_tile<false>(xh, wT, HIDDEN, QKV_N, qkv_h, nullptr, nullptr);
}

__global__ void proj_gemm_kernel(const _Float16* __restrict__ ah,
                                 const _Float16* __restrict__ wT,
                                 const float* __restrict__ bias,
                                 float* __restrict__ out) {
  gemm_tile<true>(ah, wT, HIDDEN, HIDDEN, nullptr, out, bias);
}

// ---------------------------------------------------------------------------
// 4) bias + RoPE + pack:
//    Qh [h][i][96] f16 (pre-scaled by 1/sqrt(80), cols 80..95 = 0)
//    Kh [h][i][96] f16 (cols 80..95 = 0)
//    Vt [h][d][SEQ] f16  (transposed for PV B-fragments)
// ---------------------------------------------------------------------------
__global__ void rope_pack_kernel(const _Float16* __restrict__ qkv_h,
                                 const float* __restrict__ cosE,
                                 const float* __restrict__ sinE,
                                 const float* __restrict__ b_qkv,
                                 _Float16* __restrict__ Qh,
                                 _Float16* __restrict__ Kh,
                                 _Float16* __restrict__ Vt) {
  int i = blockIdx.x, h = blockIdx.y, d = threadIdx.x;
  size_t qk_idx = ((size_t)h * SEQ + i) * HDIM_PAD + d;
  if (d >= HDIM) {
    Qh[qk_idx] = (_Float16)0.0f;
    Kh[qk_idx] = (_Float16)0.0f;
    return;
  }
  const _Float16* row = qkv_h + (size_t)i * QKV_N;
  int col = h * HDIM + d;
  float q = (float)row[col] + b_qkv[col];
  float k = (float)row[HIDDEN + col] + b_qkv[HIDDEN + col];
  float v = (float)row[2 * HIDDEN + col] + b_qkv[2 * HIDDEN + col];
  int dp = (d < HDIM / 2) ? d + HDIM / 2 : d - HDIM / 2;
  int colp = h * HDIM + dp;
  float qp = (float)row[colp] + b_qkv[colp];
  float kp = (float)row[HIDDEN + colp] + b_qkv[HIDDEN + colp];
  float rotq = (d < HDIM / 2) ? -qp : qp;
  float rotk = (d < HDIM / 2) ? -kp : kp;
  float c = cosE[(size_t)i * HDIM + d];
  float s = sinE[(size_t)i * HDIM + d];
  Qh[qk_idx] = (_Float16)((q * c + rotq * s) * INV_SCALE);
  Kh[qk_idx] = (_Float16)(k * c + rotk * s);
  Vt[((size_t)h * HDIM + d) * SEQ + i] = (_Float16)v;
}

// ---------------------------------------------------------------------------
// 5) Flash attention. grid (NHEADS, SEQ/64), 128 threads (4 waves).
//    Per 64-wide j-block: the block cooperatively DMAs K (64x96 f16) and
//    V (80x64 f16) tiles into LDS via global_load_async_to_lds_b128
//    (ASYNCcnt), then each wave (16 query rows) does 12 wmma for S,
//    online softmax with 16-lane shuffle reductions, P via a wave-private
//    LDS tile (C-frag -> A-frag), 10 wmma for PV.
// ---------------------------------------------------------------------------
__global__ void attention_kernel(const _Float16* __restrict__ Qh,
                                 const _Float16* __restrict__ Kh,
                                 const _Float16* __restrict__ Vt,
                                 _Float16* __restrict__ attn_h) {
  __shared__ __align__(32) _Float16 Kt[64][HDIM_PAD];   // 12 KB
  __shared__ __align__(32) _Float16 Vtile[HDIM][64];    // 10 KB
  __shared__ __align__(32) _Float16 Pt[4][16][64];      //  8 KB

  int h = blockIdx.x, qb = blockIdx.y;
  int w = threadIdx.x >> 5;
  int lane = threadIdx.x & 31;
  int hf = lane >> 4, n = lane & 15;
  int i0 = qb * 64 + w * 16;

  const _Float16* Qp = Qh + ((size_t)h * SEQ + i0) * HDIM_PAD;
  h16 aq0 = load_a_frag(Qp + 0, HDIM_PAD);
  h16 aq1 = load_a_frag(Qp + 32, HDIM_PAD);
  h16 aq2 = load_a_frag(Qp + 64, HDIM_PAD);

  const _Float16* Kp = Kh + (size_t)h * SEQ * HDIM_PAD;
  const _Float16* Vp = Vt + (size_t)h * HDIM * SEQ;
  uint32_t ktb = (uint32_t)(uintptr_t)(&Kt[0][0]);
  uint32_t vtb = (uint32_t)(uintptr_t)(&Vtile[0][0]);

  f8 O[5] = {f8{}, f8{}, f8{}, f8{}, f8{}};
  f8 mrow, lsum;
#pragma unroll
  for (int r = 0; r < 8; ++r) { mrow[r] = -3.0e38f; lsum[r] = 0.0f; }

  for (int j0 = 0; j0 < SEQ; j0 += 64) {
    // ---- async DMA stage: K tile is one contiguous 12288B run ----
    const char* ksrc = (const char*)(Kp + (size_t)j0 * HDIM_PAD);
#pragma unroll
    for (int it = 0; it < 6; ++it) {
      uint32_t off = (uint32_t)(threadIdx.x + it * 128) * 16;
      async_copy_b128(ktb + off, ksrc + off);
    }
    // ---- V tile: 80 rows x 128B, global stride SEQ*2 ----
#pragma unroll
    for (int it = 0; it < 5; ++it) {
      int c = threadIdx.x + it * 128;       // 640 chunks of 16B
      int d = c >> 3, part = c & 7;
      async_copy_b128(vtb + (uint32_t)(d * 128 + part * 16),
                      (const char*)(Vp + (size_t)d * SEQ + j0) + part * 16);
    }
    wait_async0();
    __syncthreads();

    // ---- S = Q K^T for 64 columns (4 col-tiles x 3 K-steps) ----
    f8 S[4] = {f8{}, f8{}, f8{}, f8{}};
#pragma unroll
    for (int jt = 0; jt < 4; ++jt) {
      S[jt] = wmma_f16(aq0, load_b_frag(&Kt[jt * 16][0], HDIM_PAD), S[jt]);
      S[jt] = wmma_f16(aq1, load_b_frag(&Kt[jt * 16][32], HDIM_PAD), S[jt]);
      S[jt] = wmma_f16(aq2, load_b_frag(&Kt[jt * 16][64], HDIM_PAD), S[jt]);
    }

    // ---- online softmax over the 64 new columns ----
    f8 P[4];
#pragma unroll
    for (int r = 0; r < 8; ++r) {
      float t = fmaxf(fmaxf(S[0][r], S[1][r]), fmaxf(S[2][r], S[3][r]));
#pragma unroll
      for (int m = 1; m < 16; m <<= 1) t = fmaxf(t, __shfl_xor(t, m, 32));
      float mn = fmaxf(mrow[r], t);
      float sc = __expf(mrow[r] - mn);
      float p0 = __expf(S[0][r] - mn);
      float p1 = __expf(S[1][r] - mn);
      float p2 = __expf(S[2][r] - mn);
      float p3 = __expf(S[3][r] - mn);
      float rs = (p0 + p1) + (p2 + p3);
#pragma unroll
      for (int m = 1; m < 16; m <<= 1) rs += __shfl_xor(rs, m, 32);
      lsum[r] = lsum[r] * sc + rs;
      mrow[r] = mn;
      P[0][r] = p0; P[1][r] = p1; P[2][r] = p2; P[3][r] = p3;
#pragma unroll
      for (int t5 = 0; t5 < 5; ++t5) O[t5][r] *= sc;
    }

    // ---- C-frag -> row-major LDS (wave-private, no barrier needed) ----
#pragma unroll
    for (int r = 0; r < 8; ++r)
#pragma unroll
      for (int jt = 0; jt < 4; ++jt)
        Pt[w][r + 8 * hf][jt * 16 + n] = (_Float16)P[jt][r];

    h16 pa0 = load_a_frag(&Pt[w][0][0], 64);
    h16 pa1 = load_a_frag(&Pt[w][0][32], 64);
#pragma unroll
    for (int t5 = 0; t5 < 5; ++t5) {
      O[t5] = wmma_f16(pa0, load_b_frag(&Vtile[16 * t5][0], 64), O[t5]);
      O[t5] = wmma_f16(pa1, load_b_frag(&Vtile[16 * t5][32], 64), O[t5]);
    }
    __syncthreads();   // all waves done reading Kt/Vtile before restage
  }

#pragma unroll
  for (int r = 0; r < 8; ++r) {
    float inv = 1.0f / lsum[r];
    int row = i0 + r + 8 * hf;
#pragma unroll
    for (int t5 = 0; t5 < 5; ++t5)
      attn_h[(size_t)row * HIDDEN + h * HDIM + 16 * t5 + n] =
          (_Float16)(O[t5][r] * inv);
  }
}

// ---------------------------------------------------------------------------
extern "C" void kernel_launch(void* const* d_in, const int* in_sizes, int n_in,
                              void* d_out, int out_size, void* d_ws, size_t ws_size,
                              hipStream_t stream) {
  const float* x      = (const float*)d_in[0];
  const float* cosE   = (const float*)d_in[1];
  const float* sinE   = (const float*)d_in[2];
  const float* w_qkv  = (const float*)d_in[3];
  const float* b_qkv  = (const float*)d_in[4];
  const float* w_proj = (const float*)d_in[5];
  const float* b_proj = (const float*)d_in[6];
  float* out = (float*)d_out;

  char* ws = (char*)d_ws;
  size_t off = 0;
  auto carve = [&](size_t bytes) {
    void* p = ws + off;
    off += (bytes + 255) & ~(size_t)255;
    return p;
  };
  _Float16* xh     = (_Float16*)carve((size_t)SEQ * HIDDEN * 2);
  _Float16* wqkvT  = (_Float16*)carve((size_t)QKV_N * HIDDEN * 2);
  _Float16* wprojT = (_Float16*)carve((size_t)HIDDEN * HIDDEN * 2);
  _Float16* qkv_h  = (_Float16*)carve((size_t)SEQ * QKV_N * 2);
  _Float16* Qh     = (_Float16*)carve((size_t)NHEADS * SEQ * HDIM_PAD * 2);
  _Float16* Kh     = (_Float16*)carve((size_t)NHEADS * SEQ * HDIM_PAD * 2);
  _Float16* Vt     = (_Float16*)carve((size_t)NHEADS * HDIM * SEQ * 2);
  _Float16* attn_h = (_Float16*)carve((size_t)SEQ * HIDDEN * 2);

  cvt_f32_to_f16_kernel<<<2048, 256, 0, stream>>>(x, xh, SEQ * HIDDEN);
  transpose_to_f16_kernel<<<4096, 256, 0, stream>>>(w_qkv, wqkvT, HIDDEN, QKV_N);
  transpose_to_f16_kernel<<<2048, 256, 0, stream>>>(w_proj, wprojT, HIDDEN, HIDDEN);

  qkv_gemm_kernel<<<dim3(QKV_N / 64, SEQ / 128), 128, 0, stream>>>(xh, wqkvT, qkv_h);
  rope_pack_kernel<<<dim3(SEQ, NHEADS), HDIM_PAD, 0, stream>>>(
      qkv_h, cosE, sinE, b_qkv, Qh, Kh, Vt);
  attention_kernel<<<dim3(NHEADS, SEQ / 64), 128, 0, stream>>>(Qh, Kh, Vt, attn_h);
  proj_gemm_kernel<<<dim3(HIDDEN / 64, SEQ / 128), 128, 0, stream>>>(
      attn_h, wprojT, b_proj, out);
}
